// DynamicGRU_40767829574273
// MI455X (gfx1250) — compile-verified, matrix operand order checked by reading, not run on hardware
//
#include <hip/hip_runtime.h>
#include <stdint.h>

// GRU dims (fixed by the reference)
#define B_  1024
#define S_  200
#define I_  512
#define H_  512
#define G3  1536   // 3*H

typedef __attribute__((ext_vector_type(16))) __bf16        bf16x16;
typedef __attribute__((ext_vector_type(8)))  float         f32x8;
typedef __attribute__((ext_vector_type(4)))  unsigned int  u32x4;
typedef __attribute__((ext_vector_type(8)))  int           i32x8;
typedef __attribute__((ext_vector_type(4)))  int           i32x4;

// ---------------------------------------------------------------------------
// One-time f32 -> bf16 weight conversion (3 MB total, reused for 200 steps).
// ---------------------------------------------------------------------------
__global__ __launch_bounds__(256) void cvt_bf16_kernel(
    const float* __restrict__ src, __bf16* __restrict__ dst, int n4)
{
    const int i = blockIdx.x * blockDim.x + threadIdx.x;
    if (i < n4) {
        const float4 f = ((const float4*)src)[i];
        __bf16* d = dst + (size_t)i * 4;
        d[0] = (__bf16)f.x; d[1] = (__bf16)f.y;
        d[2] = (__bf16)f.z; d[3] = (__bf16)f.w;
    }
}

// ---------------------------------------------------------------------------
// TDM: issue one descriptor moving a 64(row) x 32(bf16) W tile global->LDS.
// Hardware applies LDS padding: 16 DWORDs (64 B) data + 4 DWORDs (16 B) pad
// per row  => 80 B LDS row stride, matching ldsB[..][40].
// D# built per CDNA5 ISA 8.3/8.4: count=1, type=2, data_size=1 (2 B),
// tile_dim = {32, 64}, tensor_dim0_stride = 512. Groups 2/3 zero (2-D).
// ---------------------------------------------------------------------------
__device__ __forceinline__ void tdm_load_wtile(const __bf16* gsrc, unsigned lds_off)
{
    const unsigned long long ga = (unsigned long long)(uintptr_t)gsrc;
    u32x4 g0;
    g0[0] = 1u;                                                 // count = 1
    g0[1] = lds_off;                                            // lds_addr (bytes)
    g0[2] = (unsigned)(ga & 0xFFFFFFFFu);                       // global_addr[31:0]
    g0[3] = (unsigned)((ga >> 32) & 0x1FFFFFFu) | 0x80000000u;  // [56:32] | type=2
    i32x8 g1;
    g1[0] = (1 << 16) | (1 << 20) | (3 << 22) | (3 << 25);  // 2B elems, pad 16dw/4dw
    g1[1] = (int)(512u << 16);      // tensor_dim0 = 512 (k extent)
    g1[2] = (int)(1536u << 16);     // tensor_dim1 = 1536 (rows)
    g1[3] = (int)(32u  << 16);      // tile_dim0 = 32
    g1[4] = 64;                     // tile_dim1 = 64, tile_dim2 = 0
    g1[5] = 512;                    // tensor_dim0_stride = 512
    g1[6] = 0;
    g1[7] = 0;
    const i32x4 gz4 = {0, 0, 0, 0};
    const i32x8 gz8 = {0, 0, 0, 0, 0, 0, 0, 0};
    __builtin_amdgcn_tensor_load_to_lds(g0, g1, gz4, gz4, gz8, 0);
}

// ---------------------------------------------------------------------------
// GEMM kernel: Out[m, n] = sum_k A[m, k] * W[n, k] + bias[n]
//   grid.z == 0:  A = x[:, t, :]  (row stride S_*I_), W = wih_b, Out = gi
//   grid.z == 1:  A = h           (row stride H_),    W = whh_b, Out = gh
// Block = 128 threads = 4 waves. Tile 64(M) x 64(N), BK = 32.
// W tiles streamed by the Tensor Data Mover, double-buffered in LDS.
// ---------------------------------------------------------------------------
__global__ __launch_bounds__(128) void gru_gemm_kernel(
    const float*  __restrict__ x,
    const float*  __restrict__ hbuf,
    const __bf16* __restrict__ wih_b,
    const __bf16* __restrict__ whh_b,
    const float*  __restrict__ b_ih,
    const float*  __restrict__ b_hh,
    float* __restrict__ gi,
    float* __restrict__ gh,
    int t)
{
    const int zz = blockIdx.z;           // uniform per block: EXEC stays full
    const float*  Abase;
    size_t Astride;
    const __bf16* Wb;
    const float*  bias;
    float* Out;
    if (zz == 0) {
        Abase = x + (size_t)t * I_;
        Astride = (size_t)S_ * I_;
        Wb = wih_b; bias = b_ih; Out = gi;
    } else {
        Abase = hbuf;
        Astride = (size_t)H_;
        Wb = whh_b; bias = b_hh; Out = gh;
    }

    const int tileM = blockIdx.y * 64;
    const int tileN = blockIdx.x * 64;
    const int tid   = threadIdx.x;
    const int lane  = tid & 31;
    const int wave  = tid >> 5;

    // Double-buffered W tile: [buf][n_local][k_local] bf16, padded stride 40
    // (80 B rows: 64 B data + 16 B TDM pad; 16 B aligned for ds_load_b128).
    __shared__ __bf16 ldsB[2][64][40];
    const unsigned ldsBase = (unsigned)(uintptr_t)&ldsB[0][0][0];
    const unsigned bufBytes = 64u * 80u;

    const __bf16* wtile = Wb + (size_t)tileN * I_;   // advance by +kc per chunk

    // Prologue: wave 0 issues TDM for tile 0 into buffer 0.
    if (wave == 0) tdm_load_wtile(wtile, ldsBase);

    f32x8 acc[4];
#pragma unroll
    for (int a = 0; a < 4; ++a)
#pragma unroll
        for (int i = 0; i < 8; ++i) acc[a][i] = 0.0f;

    const int m_l = lane & 15;           // row within 16 (A) / col within 16 (B)
    const int kb  = (lane >> 4) << 3;    // 0 or 8: k sub-offset per lane half

    for (int kc = 0; kc < I_; kc += 32) {
        const int cur = (kc >> 5) & 1;
        __syncthreads();   // everyone finished reading the buffer we refill next
        if (wave == 0) {
            if (kc + 32 < I_) {
                tdm_load_wtile(wtile + kc + 32, ldsBase + (unsigned)(cur ^ 1) * bufBytes);
                __builtin_amdgcn_s_wait_tensorcnt(1);   // tile 'cur' complete
            } else {
                __builtin_amdgcn_s_wait_tensorcnt(0);   // last tile complete
            }
        }
        __syncthreads();   // publish TDM-written LDS to all waves

        // --- A fragment (16x32 bf16, ISA wave32 layout):
        // lane l: m = l%16; elems[0..7] = k kc+kb+0..7, elems[8..15] = kc+kb+16..23
        const float* ap = Abase + (size_t)(tileM + wave * 16 + m_l) * Astride + kc + kb;
        float4 a0 = *(const float4*)(ap);
        float4 a1 = *(const float4*)(ap + 4);
        float4 a2 = *(const float4*)(ap + 16);
        float4 a3 = *(const float4*)(ap + 20);
        if (kc + 32 < I_) __builtin_prefetch(ap + 32, 0, 3);  // global_prefetch_b8
        bf16x16 va;
        va[0]  = (__bf16)a0.x; va[1]  = (__bf16)a0.y;
        va[2]  = (__bf16)a0.z; va[3]  = (__bf16)a0.w;
        va[4]  = (__bf16)a1.x; va[5]  = (__bf16)a1.y;
        va[6]  = (__bf16)a1.z; va[7]  = (__bf16)a1.w;
        va[8]  = (__bf16)a2.x; va[9]  = (__bf16)a2.y;
        va[10] = (__bf16)a2.z; va[11] = (__bf16)a2.w;
        va[12] = (__bf16)a3.x; va[13] = (__bf16)a3.y;
        va[14] = (__bf16)a3.z; va[15] = (__bf16)a3.w;

        // --- 4 B fragments (32x16 bf16, symmetric layout over n) + WMMA
#pragma unroll
        for (int a = 0; a < 4; ++a) {
            const __bf16* bp = &ldsB[cur][a * 16 + m_l][kb];
            bf16x16 vb;
#pragma unroll
            for (int i = 0; i < 8; ++i) { vb[i] = bp[i]; vb[8 + i] = bp[16 + i]; }
            acc[a] = __builtin_amdgcn_wmma_f32_16x16x32_bf16(
                /*neg_a=*/false, va, /*neg_b=*/false, vb,
                /*c_mod=*/(short)0, acc[a],
                /*reuse_a=*/false, /*reuse_b=*/false);
        }
    }

    // Epilogue: C layout — lane l, VGPR r -> row r + 8*(l/16), col l%16
    const int rbase = (lane >> 4) << 3;
#pragma unroll
    for (int a = 0; a < 4; ++a) {
        const int n_g = tileN + a * 16 + m_l;
        const float bv = bias[n_g];
        float* op = Out + (size_t)(tileM + wave * 16 + rbase) * G3 + n_g;
#pragma unroll
        for (int r = 0; r < 8; ++r)
            op[(size_t)r * G3] = acc[a][r] + bv;
    }
}

// ---------------------------------------------------------------------------
// Elementwise GRU gate update for one timestep.
// ---------------------------------------------------------------------------
__global__ __launch_bounds__(256) void gru_elem_kernel(
    const float* __restrict__ gi,
    const float* __restrict__ gh,
    float* __restrict__ h,
    float* __restrict__ out,       // [B, S, H]
    float* __restrict__ hlast,     // nullptr except at t == S-1
    int t)
{
    const int idx = blockIdx.x * blockDim.x + threadIdx.x;   // 0 .. B*H-1
    const int b = idx >> 9;          // /512
    const int j = idx & (H_ - 1);    // %512
    const size_t g = (size_t)b * G3;

    const float ir  = gi[g + j];
    const float ii  = gi[g + H_ + j];
    const float in_ = gi[g + 2 * H_ + j];
    const float hr  = gh[g + j];
    const float hi  = gh[g + H_ + j];
    const float hn  = gh[g + 2 * H_ + j];

    const float r = 1.0f / (1.0f + expf(-(ir + hr)));
    const float z = 1.0f / (1.0f + expf(-(ii + hi)));
    const float n = tanhf(in_ + r * hn);
    const float hv = h[idx];
    const float hnew = n + z * (hv - n);

    h[idx] = hnew;
    out[(size_t)b * ((size_t)S_ * H_) + (size_t)t * H_ + j] = hnew;
    if (hlast) hlast[idx] = hnew;
}

__global__ __launch_bounds__(256) void zero_kernel(float* __restrict__ p, int n)
{
    const int idx = blockIdx.x * blockDim.x + threadIdx.x;
    if (idx < n) p[idx] = 0.0f;
}

// ---------------------------------------------------------------------------
extern "C" void kernel_launch(void* const* d_in, const int* in_sizes, int n_in,
                              void* d_out, int out_size, void* d_ws, size_t ws_size,
                              hipStream_t stream)
{
    const float* x    = (const float*)d_in[0];   // [B, S, I]
    const float* w_ih = (const float*)d_in[1];   // [3H, I]
    const float* w_hh = (const float*)d_in[2];   // [3H, H]
    const float* b_ih = (const float*)d_in[3];   // [3H]
    const float* b_hh = (const float*)d_in[4];   // [3H]

    float* out   = (float*)d_out;                         // outs [B,S,H] ...
    float* hlast = out + (size_t)B_ * S_ * H_;            // ... then h_last [B,H]

    // Workspace: h [B,H] f32 | gi [B,3H] f32 | gh [B,3H] f32 | W bf16 copies
    float* h  = (float*)d_ws;
    float* gi = h + (size_t)B_ * H_;
    float* gh = gi + (size_t)B_ * G3;
    __bf16* wih_b = (__bf16*)(gh + (size_t)B_ * G3);
    __bf16* whh_b = wih_b + (size_t)G3 * I_;

    // Weights -> bf16 once per call (inputs never mutated; deterministic).
    {
        const int n4 = (G3 * I_) / 4;   // 196608 float4s per weight matrix
        cvt_bf16_kernel<<<(n4 + 255) / 256, 256, 0, stream>>>(w_ih, wih_b, n4);
        cvt_bf16_kernel<<<(n4 + 255) / 256, 256, 0, stream>>>(w_hh, whh_b, n4);
    }

    // h0 = 0 (ws is not re-poisoned between replays; re-init every call)
    {
        const int n = B_ * H_;
        zero_kernel<<<(n + 255) / 256, 256, 0, stream>>>(h, n);
    }

    const dim3 gemm_grid(G3 / 64, B_ / 64, 2);   // 24 x 16 x 2
    const dim3 gemm_block(128);
    const int elem_n = B_ * H_;

    for (int t = 0; t < S_; ++t) {
        gru_gemm_kernel<<<gemm_grid, gemm_block, 0, stream>>>(
            x, h, wih_b, whh_b, b_ih, b_hh, gi, gh, t);
        gru_elem_kernel<<<(elem_n + 255) / 256, 256, 0, stream>>>(
            gi, gh, h, out, (t == S_ - 1) ? hlast : nullptr, t);
    }
}
